// ResNet_Mano_77549929497181
// MI455X (gfx1250) — matrix operand order repaired and checked by reading, not compiled, not run on hardware
//
#include <hip/hip_runtime.h>
#include <hip/hip_bf16.h>
#include <math.h>

typedef __attribute__((ext_vector_type(16))) _Float16 v16h;
typedef __attribute__((ext_vector_type(8)))  _Float16 v8h;
typedef __attribute__((ext_vector_type(4)))  _Float16 v4h;
typedef __attribute__((ext_vector_type(8)))  float    v8f;

#define MESH 778

// ---------------------------------------------------------------------------
// Implicit-GEMM conv + BN (+residual) (+ReLU), f16 WMMA, f32 accumulate.
// GEMM view: M=Cout, N=batch*Ho*Wo, K=Cin*KH*KW.
// Workgroup = 128 threads (4 waves). Tile: 64 (Cout) x 32 (pixels), K stage 64.
// - Double-buffered LDS, 1 barrier per stage; each stage = 2 sub-K steps
//   -> 4 WMMA per wave per barrier, 12 ds_load_b128 per wave per stage.
// - Uniform fast/slow staging split: stages fully inside K use an unguarded
//   vectorized path (only the 7x7 stem's ragged last stage takes guards).
// - B tile transposed [pixel][k] so fragments are contiguous 32B LDS reads;
//   rows padded to 72 halves (144B, 16B-aligned).
// ---------------------------------------------------------------------------
template <int KHt, int KWt>
__global__ void __launch_bounds__(128)
conv_bn_wmma(const float* __restrict__ in, const float* __restrict__ wgt,
             const float* __restrict__ bng, const float* __restrict__ bnb,
             const float* __restrict__ bnm, const float* __restrict__ bnv,
             const float* __restrict__ residual, float* __restrict__ out,
             int N, int Cin, int Hin, int Win,
             int Cout, int Hout, int Wout,
             int stride, int pad, int doRelu)
{
    constexpr int KHW = KHt * KWt;
    __shared__ alignas(16) _Float16 As[2][64][72];   // [buf][co_local][k]
    __shared__ alignas(16) _Float16 Bs[2][32][72];   // [buf][pix_local][k] (transposed)

    const int tid  = threadIdx.x;
    const int lane = tid & 31;
    const int wave = tid >> 5;

    const int pixBase = blockIdx.x * 32;
    const int coBase  = blockIdx.y * 64;
    const int Npix = N * Hout * Wout;
    const int Ktot = Cin * KHW;
    const int HWo  = Hout * Wout;
    const int HWi  = Hin * Win;

    // ---- per-thread invariants (no division inside the K loop) ----
    const int nloc = tid & 31;            // B-tile pixel row this thread fills
    const int grp  = tid >> 5;            // 0..3
    int hi0 = 0, wi0 = 0;
    const float* inp = in;
    bool pvalid = false;
    {
        int p = pixBase + nloc;
        if (p < Npix) {
            pvalid = true;
            int pn = p / HWo;
            int r  = p - pn * HWo;
            int ho = r / Wout;
            int wo = r - ho * Wout;
            hi0 = ho * stride - pad;
            wi0 = wo * stride - pad;
            inp = in + (size_t)pn * Cin * HWi + hi0 * Win + wi0;
        }
    }

    // ---- staging helpers ----
    auto stageB = [&](int buf, int k0, bool kguard) {
        #pragma unroll
        for (int e = 0; e < 16; ++e) {
            int krow = grp + e * 4;            // 0..63
            int kk = k0 + krow;
            float val = 0.0f;
            if (pvalid && (!kguard || kk < Ktot)) {
                int ci = kk / KHW;             // constant divisor
                int rr = kk - ci * KHW;
                int kh = rr / KWt;             // constant divisor
                int kw = rr - kh * KWt;
                int hi = hi0 + kh;
                int wi = wi0 + kw;
                if (hi >= 0 && hi < Hin && wi >= 0 && wi < Win)
                    val = inp[ci * HWi + kh * Win + kw];
            }
            Bs[buf][nloc][krow] = (_Float16)val;   // transposed store
        }
    };
    auto stageA = [&](int buf, int k0, bool kguard) {
        #pragma unroll
        for (int e = 0; e < 8; ++e) {
            int q   = e * 128 + tid;           // 0..1023 quads
            int row = q >> 4;                  // 0..63
            int k4  = (q & 15) * 4;            // 0,4,...,60
            int co  = coBase + row;
            float f0 = 0.f, f1 = 0.f, f2 = 0.f, f3 = 0.f;
            if (co < Cout) {
                const float* wp = &wgt[(size_t)co * Ktot + k0 + k4];
                if (!kguard) {                 // fast path: whole stage in-bounds
                    f0 = wp[0]; f1 = wp[1]; f2 = wp[2]; f3 = wp[3];
                } else {
                    int kk = k0 + k4;
                    if (kk     < Ktot) f0 = wp[0];
                    if (kk + 1 < Ktot) f1 = wp[1];
                    if (kk + 2 < Ktot) f2 = wp[2];
                    if (kk + 3 < Ktot) f3 = wp[3];
                }
            }
            v4h h;
            h[0] = (_Float16)f0; h[1] = (_Float16)f1;
            h[2] = (_Float16)f2; h[3] = (_Float16)f3;
            *(v4h*)&As[buf][row][k4] = h;      // aligned ds_store_b64
        }
    };
    auto stage = [&](int buf, int k0) {
        if (k0 + 64 <= Ktot) {                 // uniform branch: unguarded path
            stageA(buf, k0, false);
            stageB(buf, k0, false);
        } else {                               // ragged stage (stem only)
            stageA(buf, k0, true);
            stageB(buf, k0, true);
        }
    };

    v8f acc0 = {};
    v8f acc1 = {};

    stage(0, 0);
    __syncthreads();

    // fragment-addressing invariants
    const int m       = lane & 15;
    const int kb      = (lane < 16) ? 0 : 8;   // unused K group offset (halves)
    const int kb2     = (lane < 16) ? 0 : 16;  // B: K half offset (halves)
    const int rowBase = wave * 16;
    const int nn      = lane & 15;

    int buf = 0;
    for (int k0 = 0; k0 < Ktot; k0 += 64) {
        int nxt = k0 + 64;
        if (nxt < Ktot) {                      // stage next tile into other buffer
            stage(buf ^ 1, nxt);
            __builtin_prefetch(&wgt[(size_t)coBase * Ktot + nxt], 0, 0);
        }

        // ---- 2 sub-K steps, 4 WMMA per wave per stage ----
        #pragma unroll
        for (int ks = 0; ks < 2; ++ks) {
            const int ko = ks * 32;
            v8h alo = *(const v8h*)&As[buf][rowBase + m][ko + kb2];      // K=kb2..kb2+7
            v8h ahi = *(const v8h*)&As[buf][rowBase + m][ko + kb2 + 8];  // K=kb2+8..kb2+15
            v16h a = __builtin_shufflevector(alo, ahi,
                        0,1,2,3,4,5,6,7,8,9,10,11,12,13,14,15);
            v16h b0 = *(const v16h*)&Bs[buf][nn][ko + kb2];              // 32B contiguous
            v16h b1 = *(const v16h*)&Bs[buf][nn + 16][ko + kb2];

            acc0 = __builtin_amdgcn_wmma_f32_16x16x32_f16(
                       false, a, false, b0, (short)0, acc0, false, false);
            acc1 = __builtin_amdgcn_wmma_f32_16x16x32_f16(
                       false, a, false, b1, (short)0, acc1, false, false);
        }

        __syncthreads();
        buf ^= 1;
    }

    // ---- epilogue: BN + residual + ReLU, scatter to NCHW ----
    #pragma unroll
    for (int h = 0; h < 2; ++h) {
        int p = pixBase + h * 16 + nn;
        if (p < Npix) {
            int n  = p / HWo;
            int r  = p - n * HWo;
            int ho = r / Wout;
            int wo = r - ho * Wout;
            const v8f& acc = h ? acc1 : acc0;
            #pragma unroll
            for (int q = 0; q < 8; ++q) {
                int mm = q + ((lane >> 4) << 3);   // C layout: VGPR q -> M=q / q+8
                int co = coBase + wave * 16 + mm;
                if (co < Cout) {
                    float s  = bng[co] * rsqrtf(bnv[co] + 1e-5f);
                    float vv = acc[q] * s + (bnb[co] - bnm[co] * s);
                    size_t oidx = ((size_t)(n * Cout + co) * Hout + ho) * Wout + wo;
                    if (residual) vv += residual[oidx];
                    if (doRelu) vv = fmaxf(vv, 0.0f);
                    out[oidx] = vv;
                }
            }
        }
    }
}

// ---------------------------------------------------------------------------
// 3x3 stride-2 pad-1 max pool
// ---------------------------------------------------------------------------
__global__ void maxpool3x3s2(const float* __restrict__ in, float* __restrict__ out,
                             int N, int C, int H, int W, int Ho, int Wo)
{
    int idx = blockIdx.x * blockDim.x + threadIdx.x;
    int total = N * C * Ho * Wo;
    if (idx >= total) return;
    int wo = idx % Wo; int t = idx / Wo;
    int ho = t % Ho;   t /= Ho;
    int c  = t % C;    int n = t / C;
    float mx = -INFINITY;
    for (int kh = 0; kh < 3; ++kh)
        for (int kw = 0; kw < 3; ++kw) {
            int hi = ho * 2 - 1 + kh, wi = wo * 2 - 1 + kw;
            if (hi >= 0 && hi < H && wi >= 0 && wi < W)
                mx = fmaxf(mx, in[((size_t)(n * C + c) * H + hi) * W + wi]);
        }
    out[idx] = mx;
}

// ---------------------------------------------------------------------------
// Global average pool (512x7x7) + FC(22x512) + mean_vec  -> xs[32][22]
// ---------------------------------------------------------------------------
__global__ void __launch_bounds__(256)
avgpool_fc(const float* __restrict__ in, const float* __restrict__ fcw,
           const float* __restrict__ fcb, const float* __restrict__ meanvec,
           float* __restrict__ xs)
{
    __shared__ float cm[512];
    int b = blockIdx.x, tid = threadIdx.x;
    for (int c = tid; c < 512; c += 256) {
        const float* p = in + (size_t)(b * 512 + c) * 49;
        float s = 0.0f;
        for (int i = 0; i < 49; ++i) s += p[i];
        cm[c] = s * (1.0f / 49.0f);
    }
    __syncthreads();
    if (tid < 22) {
        float a = fcb[tid] + meanvec[tid];
        for (int c = 0; c < 512; ++c) a += fcw[tid * 512 + c] * cm[c];
        xs[b * 22 + tid] = a;
    }
}

// ---------------------------------------------------------------------------
// MANO: one workgroup per batch sample; everything lives in LDS (~32KB).
// ---------------------------------------------------------------------------
__device__ __constant__ int kParents[16]   = {0,0,1,2,0,4,5,0,7,8,0,10,11,0,13,14};
__device__ __constant__ int kJsrcJoint[21] = {0,1,2,3,-1,4,5,6,-1,7,8,9,-1,10,11,12,-1,13,14,15,-1};
__device__ __constant__ int kJsrcVert[21]  = {0,0,0,0,333,0,0,0,444,0,0,0,672,0,0,0,555,0,0,0,745};

__device__ inline void mat3mul(const float* A, const float* B, float* C) {
    for (int i = 0; i < 3; ++i)
        for (int j = 0; j < 3; ++j) {
            float s = 0.0f;
            for (int k = 0; k < 3; ++k) s += A[i*3+k] * B[k*3+j];
            C[i*3+j] = s;
        }
}

__device__ void rodrigues3(const float* r, float* R) {
    float t2 = r[0]*r[0] + r[1]*r[1] + r[2]*r[2];
    float th = sqrtf(t2);
    if (th < 1e-30f) {
        float S[9] = {0,-r[2],r[1], r[2],0,-r[0], -r[1],r[0],0};
        float S2[9]; mat3mul(S, S, S2);
        float c1 = 1.0f - t2 / 6.0f, c2 = 0.5f - t2 / 24.0f;
        for (int i = 0; i < 9; ++i) R[i] = c1 * S[i] + c2 * S2[i];
    } else {
        float inv = 1.0f / th;
        float n0 = r[0]*inv, n1 = r[1]*inv, n2 = r[2]*inv;
        float Sn[9] = {0,-n2,n1, n2,0,-n0, -n1,n0,0};
        float Sn2[9]; mat3mul(Sn, Sn, Sn2);
        float s = sinf(th), c = 1.0f - cosf(th);
        for (int i = 0; i < 9; ++i) R[i] = s * Sn[i] + c * Sn2[i];
    }
    R[0] += 1.0f; R[4] += 1.0f; R[8] += 1.0f;
}

__global__ void __launch_bounds__(256)
mano_kernel(const float* __restrict__ xs,
            const float* __restrict__ jreg,     // 16 x 778
            const float* __restrict__ hcomp,    // 6 x 45
            const float* __restrict__ hmean,    // 45
            const float* __restrict__ posedirs, // 778*3 x 135
            const float* __restrict__ shapedirs,// 778*3 x 10
            const float* __restrict__ vtempl,   // 778*3
            const float* __restrict__ wts,      // 778 x 16
            float* __restrict__ out2d, float* __restrict__ out3d)
{
    const int b = blockIdx.x;
    const int tid = threadIdx.x;
    __shared__ float sc, tr0, tr1;
    __shared__ float rot[3], theta[6], beta[10];
    __shared__ float poses[48];         // 16 x 3
    __shared__ float Rs[16*9];
    __shared__ float vshaped[MESH*3];
    __shared__ float vposed[MESH*3];
    __shared__ float Jm[48];            // 16 x 3
    __shared__ float posew[135];
    __shared__ float G[16*16];
    __shared__ float Am[16*16];
    __shared__ float vsk[MESH*3];
    __shared__ float Jtr[21*3];
    __shared__ float Rr[9];

    if (tid == 0) { sc = xs[b*22+0]; tr0 = xs[b*22+1]; tr1 = xs[b*22+2]; }
    if (tid < 3)  rot[tid]   = xs[b*22+3+tid];
    if (tid < 6)  theta[tid] = xs[b*22+6+tid];
    if (tid < 10) beta[tid]  = xs[b*22+12+tid];
    __syncthreads();

    // poses = [root(pi,0,0), hands_mean + theta @ hands_components]
    if (tid < 48) {
        if (tid < 3) poses[tid] = (tid == 0) ? 3.14159265358979f : 0.0f;
        else {
            int i = tid - 3;
            float s = hmean[i];
            for (int k = 0; k < 6; ++k) s += theta[k] * hcomp[k*45 + i];
            poses[tid] = s;
        }
    }
    __syncthreads();

    if (tid < 16) rodrigues3(&poses[tid*3], &Rs[tid*9]);
    // v_shaped = v_template + shapedirs @ beta
    for (int i = tid; i < MESH*3; i += 256) {
        float s = vtempl[i];
        for (int k = 0; k < 10; ++k) s += beta[k] * shapedirs[(size_t)i*10 + k];
        vshaped[i] = s;
    }
    __syncthreads();

    // J = J_regressor @ v_shaped ; posew = (Rs[1:]-I).flatten()
    if (tid < 48) {
        int j = tid / 3, c = tid % 3;
        float s = 0.0f;
        for (int v = 0; v < MESH; ++v) s += jreg[j*MESH + v] * vshaped[v*3 + c];
        Jm[tid] = s;
    }
    if (tid >= 64 && tid < 64 + 135) {
        int i = tid - 64;
        int jj = i / 9 + 1, e = i % 9;
        float val = Rs[jj*9 + e];
        if (e == 0 || e == 4 || e == 8) val -= 1.0f;
        posew[i] = val;
    }
    __syncthreads();

    // v_posed = v_shaped + posedirs @ posew
    for (int i = tid; i < MESH*3; i += 256) {
        float s = vshaped[i];
        const float* pd = &posedirs[(size_t)i * 135];
        for (int k = 0; k < 135; ++k) s += pd[k] * posew[k];
        vposed[i] = s;
    }
    __syncthreads();

    // kinematic chain (serial, 16 joints)
    if (tid == 0) {
        float* g0 = &G[0];
        for (int p = 0; p < 3; ++p) {
            for (int q = 0; q < 3; ++q) g0[p*4+q] = Rs[p*3+q];
            g0[p*4+3] = Jm[p];
        }
        g0[12]=0; g0[13]=0; g0[14]=0; g0[15]=1;
        for (int i = 1; i < 16; ++i) {
            int par = kParents[i];
            float L[16];
            for (int p = 0; p < 3; ++p) {
                for (int q = 0; q < 3; ++q) L[p*4+q] = Rs[i*9 + p*3 + q];
                L[p*4+3] = Jm[i*3+p] - Jm[par*3+p];
            }
            L[12]=0; L[13]=0; L[14]=0; L[15]=1;
            const float* gp = &G[par*16];
            float* gi = &G[i*16];
            for (int p = 0; p < 4; ++p)
                for (int q = 0; q < 4; ++q) {
                    float s = 0.0f;
                    for (int k = 0; k < 4; ++k) s += gp[p*4+k] * L[k*4+q];
                    gi[p*4+q] = s;
                }
        }
    }
    __syncthreads();

    // A = G - pad(corr): subtract G[:, :3]@J from column 3
    if (tid < 256) {
        int j = tid >> 4, e = tid & 15, p = e >> 2, q = e & 3;
        float val = G[tid];
        if (q == 3 && p < 3) {
            float corr = 0.0f;
            for (int k = 0; k < 3; ++k) corr += G[j*16 + p*4 + k] * Jm[j*3 + k];
            val -= corr;
        }
        Am[tid] = val;
    }
    __syncthreads();

    // linear blend skinning
    for (int v = tid; v < MESH; v += 256) {
        float T[12];
        for (int e = 0; e < 12; ++e) {
            float s = 0.0f;
            for (int j = 0; j < 16; ++j) s += wts[v*16 + j] * Am[j*16 + e];
            T[e] = s;
        }
        float x = vposed[v*3], y = vposed[v*3+1], z = vposed[v*3+2];
        vsk[v*3+0] = T[0]*x + T[1]*y + T[2]*z  + T[3];
        vsk[v*3+1] = T[4]*x + T[5]*y + T[6]*z  + T[7];
        vsk[v*3+2] = T[8]*x + T[9]*y + T[10]*z + T[11];
    }
    if (tid == 0) rodrigues3(rot, Rr);
    __syncthreads();

    // Jtr with fingertip inserts
    if (tid < 63) {
        int j = tid / 3, c = tid % 3;
        int js = kJsrcJoint[j];
        Jtr[tid] = (js >= 0) ? G[js*16 + c*4 + 3] : vsk[kJsrcVert[j]*3 + c];
    }
    __syncthreads();

    // global rotation + projection; x3d = [J_out(21); v_out(778)]
    for (int i = tid; i < 799; i += 256) {
        const float* src = (i < 21) ? &Jtr[i*3] : &vsk[(i-21)*3];
        float x = src[0], y = src[1], z = src[2];
        float p0 = Rr[0]*x + Rr[1]*y + Rr[2]*z;
        float p1 = Rr[3]*x + Rr[4]*y + Rr[5]*z;
        float p2 = Rr[6]*x + Rr[7]*y + Rr[8]*z;
        size_t o3 = (size_t)b*799*3 + (size_t)i*3;
        out3d[o3] = p0; out3d[o3+1] = p1; out3d[o3+2] = p2;
        size_t o2 = (size_t)b*1598 + (size_t)i*2;
        out2d[o2]   = tr0 + sc * p0;
        out2d[o2+1] = tr1 + sc * p1;
    }
}

// ---------------------------------------------------------------------------
// Host orchestration
// ---------------------------------------------------------------------------
struct BP {
    const float* c1; const float* b1[4];
    const float* c2; const float* b2[4];
    const float* dc; const float* db[4];
    bool down;
};

extern "C" void kernel_launch(void* const* d_in, const int* in_sizes, int n_in,
                              void* d_out, int out_size, void* d_ws, size_t ws_size,
                              hipStream_t stream)
{
    (void)in_sizes; (void)out_size; (void)ws_size;
    auto F = [&](int i) { return (const float*)d_in[i]; };

    // --- consts (jax tree order: sorted dict keys) ---
    const float* J_reg   = F(0);   // 16x778
    const float* hcomp   = F(1);   // 6x45
    const float* hmean   = F(2);   // 45
    const float* meanvec = F(3);   // 22
    const float* posed   = F(4);   // 778*3*135
    const float* shaped  = F(5);   // 778*3*10
    const float* vtempl  = F(6);   // 778*3
    const float* wts     = F(7);   // 778*16
    // --- params: bn1(4), conv1, fc_b, fc_w, layers ---
    const float* bn1[4]  = {F(8), F(9), F(10), F(11)};
    const float* conv1   = F(12);
    const float* fcb     = F(13);
    const float* fcw     = F(14);
    const float* x       = F(n_in - 1);  // 32x3x224x224

    BP blk[8];
    int  bases[8] = {15, 25, 35, 50, 60, 75, 85, 100};
    bool hasd[8]  = {false, false, true, false, true, false, true, false};
    for (int i = 0; i < 8; ++i) {
        int base = bases[i];
        for (int k = 0; k < 4; ++k) blk[i].b1[k] = F(base + k);
        for (int k = 0; k < 4; ++k) blk[i].b2[k] = F(base + 4 + k);
        blk[i].c1 = F(base + 8);
        blk[i].c2 = F(base + 9);
        blk[i].down = hasd[i];
        if (hasd[i]) {
            for (int k = 0; k < 4; ++k) blk[i].db[k] = F(base + 10 + k);
            blk[i].dc = F(base + 14);
        } else {
            blk[i].dc = nullptr;
            for (int k = 0; k < 4; ++k) blk[i].db[k] = nullptr;
        }
    }

    // --- workspace layout (floats) ---
    float* WS = (float*)d_ws;
    float* S    = WS;                        // 32*64*112*112 = 25,690,112 (stem; reused for downsample)
    float* act0 = S    + 25690112;           // 32*64*56*56 = 6,422,528 each
    float* act1 = act0 + 6422528;
    float* act2 = act1 + 6422528;
    float* xs   = act2 + 6422528;            // 32*22
    float* act[3] = {act0, act1, act2};

    auto launch_conv = [&](const float* in, const float* w, const float* const* bn,
                           const float* res, float* out,
                           int N, int Cin, int Hin, int Win,
                           int Cout, int Hout, int Wout,
                           int KH, int KW, int stride, int pad, int relu) {
        int Npix = N * Hout * Wout;
        dim3 grid((Npix + 31) / 32, (Cout + 63) / 64);
        if (KH == 7 && KW == 7)
            conv_bn_wmma<7,7><<<grid, 128, 0, stream>>>(in, w, bn[0], bn[1], bn[2], bn[3],
                res, out, N, Cin, Hin, Win, Cout, Hout, Wout, stride, pad, relu);
        else if (KH == 3 && KW == 3)
            conv_bn_wmma<3,3><<<grid, 128, 0, stream>>>(in, w, bn[0], bn[1], bn[2], bn[3],
                res, out, N, Cin, Hin, Win, Cout, Hout, Wout, stride, pad, relu);
        else
            conv_bn_wmma<1,1><<<grid, 128, 0, stream>>>(in, w, bn[0], bn[1], bn[2], bn[3],
                res, out, N, Cin, Hin, Win, Cout, Hout, Wout, stride, pad, relu);
    };

    // stem: conv7x7 s2 p3 + bn + relu -> 64x112x112
    launch_conv(x, conv1, bn1, nullptr, S, 32, 3, 224, 224, 64, 112, 112, 7, 7, 2, 3, 1);
    // maxpool 3x3 s2 p1 -> 64x56x56
    {
        int total = 32 * 64 * 56 * 56;
        maxpool3x3s2<<<(total + 255) / 256, 256, 0, stream>>>(S, act[0], 32, 64, 112, 112, 56, 56);
    }

    int planes[4]  = {64, 128, 256, 512};
    int strides[4] = {1, 2, 2, 2};
    int inC = 64, inH = 56, inW = 56;
    int cur = 0, bi = 0;
    for (int li = 0; li < 4; ++li) {
        for (int bb = 0; bb < 2; ++bb) {
            const BP& P = blk[bi++];
            int s  = (bb == 0) ? strides[li] : 1;
            int oc = planes[li];
            int oH = (inH + 2 - 3) / s + 1;
            int oW = (inW + 2 - 3) / s + 1;
            float* y = act[(cur + 1) % 3];
            float* z = act[(cur + 2) % 3];
            // conv1 + bn1 + relu
            launch_conv(act[cur], P.c1, P.b1, nullptr, y,
                        32, inC, inH, inW, oc, oH, oW, 3, 3, s, 1, 1);
            const float* idt = act[cur];
            if (P.down) {  // 1x1 s-strided downsample + bn (no relu) into stem buffer
                launch_conv(act[cur], P.dc, P.db, nullptr, S,
                            32, inC, inH, inW, oc, oH, oW, 1, 1, s, 0, 0);
                idt = S;
            }
            // conv2 + bn2 + residual + relu
            launch_conv(y, P.c2, P.b2, idt, z,
                        32, oc, oH, oW, oc, oH, oW, 3, 3, 1, 1, 1);
            cur = (cur + 2) % 3;
            inC = oc; inH = oH; inW = oW;
        }
    }

    // avgpool + FC + mean_vec -> xs[32][22]
    avgpool_fc<<<32, 256, 0, stream>>>(act[cur], fcw, fcb, meanvec, xs);

    // MANO -> d_out = [x2d (32*1598) | x3d (32*799*3)]
    float* out2d = (float*)d_out;
    float* out3d = out2d + 32 * 1598;
    mano_kernel<<<32, 256, 0, stream>>>(xs, J_reg, hcomp, hmean, posed, shaped,
                                        vtempl, wts, out2d, out3d);
}
